// ComnetModel_60713657696781
// MI455X (gfx1250) — compile-verified, matrix operand order checked by reading, not run on hardware
//
#include <hip/hip_runtime.h>
#include <hip/hip_bf16.h>

// ---------------------------------------------------------------------------
// Types for CDNA5 WMMA (wave32): v_wmma_f32_16x16x32_bf16
// ---------------------------------------------------------------------------
typedef __attribute__((ext_vector_type(16))) __bf16 bf16x16;
typedef __attribute__((ext_vector_type(8)))  __bf16 bf16x8;
typedef __attribute__((ext_vector_type(4)))  __bf16 bf16x4;
typedef __attribute__((ext_vector_type(8)))  float  v8f;

#define WMMA_BF16(A, B, C) \
  __builtin_amdgcn_wmma_f32_16x16x32_bf16(false, (A), false, (B), (short)0, (C), false, false)

// Constants of the model (fixed by the reference).
#define N_LINK 100000
#define N_PATH 100000
#define E_EDGE 600000
#define D_DIM  128
#define H_MSG  256
#define T_IT   4

// LDS row strides (elements): pad by 8 bf16 (16B) so successive rows rotate by
// 4 LDS banks -> conflict-free ds_load_b128 fragment reads across lanes 0..15.
#define A_STR  264   // 256 + 8
#define S_STR  136   // 128 + 8

// ---------------------------------------------------------------------------
// Load one 16x32 bf16 WMMA operand fragment (A layout, or B as B^T rows).
// lane (0..31): row = caller passes m or n = lane&15,
// K per lane:   {k0 + 8*(lane>>4) .. +7} and {k0 + 16 + 8*(lane>>4) .. +7}
// Both runs are 16-byte aligned for all strides used here (multiples of 8).
// ---------------------------------------------------------------------------
static __device__ __forceinline__
bf16x16 ldfrag(const __bf16* __restrict__ base, int stride, int row, int k0, int lane) {
  const int g = (lane >> 4) << 3;
  union { bf16x16 v; bf16x8 h[2]; } f;
  const __bf16* p = base + (long)row * stride + k0 + g;
  f.h[0] = *(const bf16x8*)(p);
  f.h[1] = *(const bf16x8*)(p + 16);
  return f.v;
}

// ---------------------------------------------------------------------------
// Utility kernels
// ---------------------------------------------------------------------------
// Transpose-convert fp32 src[K][N] (row-major) -> bf16 dst[N][K]
__global__ void k_tconv(const float* __restrict__ src, __bf16* __restrict__ dst,
                        int K, int N) {
  long i = (long)blockIdx.x * blockDim.x + threadIdx.x;
  long tot = (long)K * N;
  if (i < tot) {
    int k = (int)(i / N), n = (int)(i % N);
    dst[(long)n * K + k] = (__bf16)src[i];
  }
}

// Straight fp32 -> bf16 convert
__global__ void k_conv(const float* __restrict__ src, __bf16* __restrict__ dst, long tot) {
  for (long i = (long)blockIdx.x * blockDim.x + threadIdx.x; i < tot;
       i += (long)gridDim.x * blockDim.x)
    dst[i] = (__bf16)src[i];
}

__global__ void k_zero(float* __restrict__ p, long tot) {
  for (long i = (long)blockIdx.x * blockDim.x + threadIdx.x; i < tot;
       i += (long)gridDim.x * blockDim.x)
    p[i] = 0.0f;
}

// ---------------------------------------------------------------------------
// Fused per-edge MLP + segment-sum scatter.
// Block = 32 edges (2 M-tiles of 16), 256 threads = 8 waves.
//   A = concat(link[src], path[dst]) : 32 x 256 bf16 in LDS
//   H = relu(A @ W1 + b1)            : 32 x 256 bf16 in LDS
//   msg = H @ W2 + b2                : scattered with global_atomic_add_f32
// Weights are bf16, pre-transposed to [N][K] so each lane reads its column's
// row contiguously (stays hot in 192MB L2). Layer 1 loads each A fragment
// ONCE per K-chunk and feeds 4 accumulators (2 N-tiles x 2 M-tiles).
// ---------------------------------------------------------------------------
__global__ __launch_bounds__(256)
void k_edge_mlp(const __bf16* __restrict__ linkb, const __bf16* __restrict__ pathb,
                const int* __restrict__ src, const int* __restrict__ dst,
                const __bf16* __restrict__ W1t, const float* __restrict__ b1,
                const __bf16* __restrict__ W2t, const float* __restrict__ b2,
                float* __restrict__ agg, int ne) {
  __shared__ __bf16 Asm[32 * A_STR];
  __shared__ __bf16 Hsm[32 * A_STR];
  const int tid  = threadIdx.x;
  const int lane = tid & 31;
  const int wv   = tid >> 5;
  const int nn   = lane & 15;
  const int mhi  = (lane >> 4) * 8;
  const int e0   = blockIdx.x * 32;

  // Gather: 32 rows x 256 cols as 8-byte chunks (2048 chunks / 256 threads).
  for (int i = 0; i < 8; ++i) {
    int c    = tid + i * 256;
    int row  = c >> 6;            // 64 chunks per row
    int col4 = (c & 63) * 4;
    int e    = e0 + row;
    bf16x4 v = {};
    if (e < ne) {
      if (col4 < D_DIM) v = *(const bf16x4*)(linkb + (long)src[e] * D_DIM + col4);
      else              v = *(const bf16x4*)(pathb + (long)dst[e] * D_DIM + (col4 - D_DIM));
    }
    *(bf16x4*)(Asm + row * A_STR + col4) = v;
  }
  __syncthreads();

  // Layer 1: wave w owns N-tiles {w, w+8}; A fragments loaded once per K-chunk.
  {
    const int n0 = wv * 16;
    const int n1 = (wv + 8) * 16;
    v8f acc00 = {}; v8f acc01 = {}; v8f acc10 = {}; v8f acc11 = {};
    for (int k0 = 0; k0 < 2 * D_DIM; k0 += 32) {
      bf16x16 a0 = ldfrag(Asm, A_STR, nn,      k0, lane);
      bf16x16 a1 = ldfrag(Asm, A_STR, 16 + nn, k0, lane);
      bf16x16 w0 = ldfrag(W1t, 2 * D_DIM, n0 + nn, k0, lane);
      acc00 = WMMA_BF16(a0, w0, acc00);
      acc01 = WMMA_BF16(a1, w0, acc01);
      bf16x16 w1 = ldfrag(W1t, 2 * D_DIM, n1 + nn, k0, lane);
      acc10 = WMMA_BF16(a0, w1, acc10);
      acc11 = WMMA_BF16(a1, w1, acc11);
    }
    const float bb0 = b1[n0 + nn];
    const float bb1 = b1[n1 + nn];
#pragma unroll
    for (int r = 0; r < 8; ++r) {
      float v00 = acc00[r] + bb0; v00 = v00 > 0.0f ? v00 : 0.0f;
      float v01 = acc01[r] + bb0; v01 = v01 > 0.0f ? v01 : 0.0f;
      float v10 = acc10[r] + bb1; v10 = v10 > 0.0f ? v10 : 0.0f;
      float v11 = acc11[r] + bb1; v11 = v11 > 0.0f ? v11 : 0.0f;
      Hsm[(r + mhi) * A_STR + n0 + nn]      = (__bf16)v00;
      Hsm[(16 + r + mhi) * A_STR + n0 + nn] = (__bf16)v01;
      Hsm[(r + mhi) * A_STR + n1 + nn]      = (__bf16)v10;
      Hsm[(16 + r + mhi) * A_STR + n1 + nn] = (__bf16)v11;
    }
  }
  __syncthreads();

  // Layer 2: msg (32 x 128) = H @ W2 + b2 ; 8 N-tiles, one per wave.
  {
    int n0 = wv * 16;
    v8f m0 = {}; v8f m1 = {};
    for (int k0 = 0; k0 < H_MSG; k0 += 32) {
      bf16x16 w  = ldfrag(W2t, H_MSG, n0 + nn, k0, lane);
      bf16x16 a0 = ldfrag(Hsm, A_STR, nn,      k0, lane);
      bf16x16 a1 = ldfrag(Hsm, A_STR, 16 + nn, k0, lane);
      m0 = WMMA_BF16(a0, w, m0);
      m1 = WMMA_BF16(a1, w, m1);
    }
    float bb = b2[n0 + nn];
#pragma unroll
    for (int r = 0; r < 8; ++r) {
      int e1 = e0 + r + mhi;
      if (e1 < ne) atomicAdd(agg + (long)dst[e1] * D_DIM + n0 + nn, m0[r] + bb);
      int e2 = e0 + 16 + r + mhi;
      if (e2 < ne) atomicAdd(agg + (long)dst[e2] * D_DIM + n0 + nn, m1[r] + bb);
    }
  }
}

// ---------------------------------------------------------------------------
// GRU update: per block 16 nodes, 256 threads (8 waves).
// Wave w owns columns [w*16, w*16+16) of D; computes 6 WMMA tile-GEMMs:
// gx_{z,r,h} = agg @ gru_k  (+bias later), gh_{z,r,h} = h @ gru_rk.
// A fragments (aA/aH) are each reused by 3 WMMAs per K-chunk already.
// ---------------------------------------------------------------------------
__global__ __launch_bounds__(256)
void k_gru(const float* __restrict__ agg, const float* __restrict__ hin,
           const __bf16* __restrict__ gkt, const __bf16* __restrict__ grkt,
           const float* __restrict__ gb, float* __restrict__ hout) {
  __shared__ __bf16 Aag[16 * S_STR];
  __shared__ __bf16 Ah [16 * S_STR];
  const int tid  = threadIdx.x;
  const int lane = tid & 31;
  const int wv   = tid >> 5;
  const int nn   = lane & 15;
  const int mhi  = (lane >> 4) * 8;
  const long base = (long)blockIdx.x * 16;

  // Stage & convert the 16x128 agg and h tiles to bf16 in LDS.
  for (int i = 0; i < 8; ++i) {
    int idx = tid + i * 256;
    int rr = idx >> 7, cc = idx & 127;
    Aag[rr * S_STR + cc] = (__bf16)agg[(base + rr) * D_DIM + cc];
    Ah [rr * S_STR + cc] = (__bf16)hin[(base + rr) * D_DIM + cc];
  }
  __syncthreads();

  v8f gxz = {}, gxr = {}, gxh = {}, ghz = {}, ghr = {}, ghh = {};
  for (int k0 = 0; k0 < D_DIM; k0 += 32) {
    bf16x16 aA = ldfrag(Aag, S_STR, nn, k0, lane);
    bf16x16 aH = ldfrag(Ah,  S_STR, nn, k0, lane);
    bf16x16 w;
    w = ldfrag(gkt,  D_DIM,           wv * 16 + nn, k0, lane); gxz = WMMA_BF16(aA, w, gxz);
    w = ldfrag(gkt,  D_DIM, D_DIM   + wv * 16 + nn, k0, lane); gxr = WMMA_BF16(aA, w, gxr);
    w = ldfrag(gkt,  D_DIM, 2*D_DIM + wv * 16 + nn, k0, lane); gxh = WMMA_BF16(aA, w, gxh);
    w = ldfrag(grkt, D_DIM,           wv * 16 + nn, k0, lane); ghz = WMMA_BF16(aH, w, ghz);
    w = ldfrag(grkt, D_DIM, D_DIM   + wv * 16 + nn, k0, lane); ghr = WMMA_BF16(aH, w, ghr);
    w = ldfrag(grkt, D_DIM, 2*D_DIM + wv * 16 + nn, k0, lane); ghh = WMMA_BF16(aH, w, ghh);
  }

  const float bz = gb[wv * 16 + nn];
  const float br = gb[D_DIM + wv * 16 + nn];
  const float bh = gb[2 * D_DIM + wv * 16 + nn];
#pragma unroll
  for (int r = 0; r < 8; ++r) {
    long node = base + r + mhi;
    float hold = hin[node * D_DIM + wv * 16 + nn];
    float z  = 1.0f / (1.0f + __expf(-(gxz[r] + bz + ghz[r])));
    float rg = 1.0f / (1.0f + __expf(-(gxr[r] + br + ghr[r])));
    float pre = gxh[r] + bh + rg * ghh[r];
    float hh = 2.0f / (1.0f + __expf(-2.0f * pre)) - 1.0f;   // tanh
    hout[node * D_DIM + wv * 16 + nn] = z * hold + (1.0f - z) * hh;
  }
}

// ---------------------------------------------------------------------------
// Readout: out = relu(h @ R1 + rb1) @ R2 + rb2 ; 16 nodes per block.
// A fragments loaded once per K-chunk, feeding both N-tiles of the wave.
// ---------------------------------------------------------------------------
__global__ __launch_bounds__(256)
void k_readout(const float* __restrict__ hin, const __bf16* __restrict__ R1t,
               const float* __restrict__ rb1, const float* __restrict__ R2,
               const float* __restrict__ rb2, float* __restrict__ out) {
  __shared__ __bf16 Ah[16 * S_STR];
  __shared__ float  Hs[16 * 260];
  const int tid  = threadIdx.x;
  const int lane = tid & 31;
  const int wv   = tid >> 5;
  const int nn   = lane & 15;
  const int mhi  = (lane >> 4) * 8;
  const long base = (long)blockIdx.x * 16;

  for (int i = 0; i < 8; ++i) {
    int idx = tid + i * 256;
    int rr = idx >> 7, cc = idx & 127;
    Ah[rr * S_STR + cc] = (__bf16)hin[(base + rr) * D_DIM + cc];
  }
  __syncthreads();

  {
    const int n0 = wv * 16;
    const int n1 = (wv + 8) * 16;
    v8f acc0 = {}; v8f acc1 = {};
    for (int k0 = 0; k0 < D_DIM; k0 += 32) {
      bf16x16 a  = ldfrag(Ah,  S_STR, nn,      k0, lane);
      bf16x16 w0 = ldfrag(R1t, D_DIM, n0 + nn, k0, lane);
      acc0 = WMMA_BF16(a, w0, acc0);
      bf16x16 w1 = ldfrag(R1t, D_DIM, n1 + nn, k0, lane);
      acc1 = WMMA_BF16(a, w1, acc1);
    }
    const float bb0 = rb1[n0 + nn];
    const float bb1 = rb1[n1 + nn];
#pragma unroll
    for (int r = 0; r < 8; ++r) {
      float v0 = acc0[r] + bb0;
      float v1 = acc1[r] + bb1;
      Hs[(r + mhi) * 260 + n0 + nn] = v0 > 0.0f ? v0 : 0.0f;
      Hs[(r + mhi) * 260 + n1 + nn] = v1 > 0.0f ? v1 : 0.0f;
    }
  }
  __syncthreads();

  if (tid < 16) {
    float s = rb2[0];
    for (int c = 0; c < H_MSG; ++c) s += Hs[tid * 260 + c] * R2[c];
    out[base + tid] = s;
  }
}

// ---------------------------------------------------------------------------
// Host orchestration (graph-capture safe: launches only, fixed T=4 unrolled)
// ---------------------------------------------------------------------------
extern "C" void kernel_launch(void* const* d_in, const int* in_sizes, int n_in,
                              void* d_out, int out_size, void* d_ws, size_t ws_size,
                              hipStream_t stream) {
  (void)in_sizes; (void)n_in; (void)out_size; (void)ws_size;

  const float* link_x = (const float*)d_in[0];
  const float* path_x = (const float*)d_in[1];
  const int*   srcI   = (const int*)  d_in[2];
  const int*   dstI   = (const int*)  d_in[3];
  const float* W1     = (const float*)d_in[4];
  const float* b1     = (const float*)d_in[5];
  const float* W2     = (const float*)d_in[6];
  const float* b2     = (const float*)d_in[7];
  const float* gru_k  = (const float*)d_in[8];
  const float* gru_rk = (const float*)d_in[9];
  const float* gru_b  = (const float*)d_in[10];
  const float* R1     = (const float*)d_in[11];
  const float* rb1    = (const float*)d_in[12];
  const float* R2     = (const float*)d_in[13];
  const float* rb2    = (const float*)d_in[14];
  float* outp = (float*)d_out;

  // Workspace carve-up (all sizes multiples of 256B).
  char* w = (char*)d_ws;
  float* ps0   = (float*)w;  w += (long)N_PATH * D_DIM * 4;
  float* ps1   = (float*)w;  w += (long)N_PATH * D_DIM * 4;
  float* agg   = (float*)w;  w += (long)N_PATH * D_DIM * 4;
  __bf16* linkb = (__bf16*)w; w += (long)N_LINK * D_DIM * 2;
  __bf16* pathb = (__bf16*)w; w += (long)N_PATH * D_DIM * 2;
  __bf16* W1t  = (__bf16*)w; w += (long)(2 * D_DIM) * H_MSG * 2;
  __bf16* W2t  = (__bf16*)w; w += (long)H_MSG * D_DIM * 2;
  __bf16* gkt  = (__bf16*)w; w += (long)D_DIM * (3 * D_DIM) * 2;
  __bf16* grkt = (__bf16*)w; w += (long)D_DIM * (3 * D_DIM) * 2;
  __bf16* R1t  = (__bf16*)w; w += (long)D_DIM * H_MSG * 2;

  // One-time weight transpose/convert to bf16 [N][K].
  k_tconv<<<(2 * D_DIM * H_MSG + 255) / 256, 256, 0, stream>>>(W1, W1t, 2 * D_DIM, H_MSG);
  k_tconv<<<(H_MSG * D_DIM + 255) / 256, 256, 0, stream>>>(W2, W2t, H_MSG, D_DIM);
  k_tconv<<<(D_DIM * 3 * D_DIM + 255) / 256, 256, 0, stream>>>(gru_k, gkt, D_DIM, 3 * D_DIM);
  k_tconv<<<(D_DIM * 3 * D_DIM + 255) / 256, 256, 0, stream>>>(gru_rk, grkt, D_DIM, 3 * D_DIM);
  k_tconv<<<(D_DIM * H_MSG + 255) / 256, 256, 0, stream>>>(R1, R1t, D_DIM, H_MSG);
  // link states never change: convert once.
  k_conv<<<2048, 256, 0, stream>>>(link_x, linkb, (long)N_LINK * D_DIM);

  const float* cur = path_x;
  float* ping[2] = { ps0, ps1 };
  for (int it = 0; it < T_IT; ++it) {
    k_conv<<<2048, 256, 0, stream>>>(cur, pathb, (long)N_PATH * D_DIM);
    k_zero<<<2048, 256, 0, stream>>>(agg, (long)N_PATH * D_DIM);
    k_edge_mlp<<<(E_EDGE + 31) / 32, 256, 0, stream>>>(
        linkb, pathb, srcI, dstI, W1t, b1, W2t, b2, agg, E_EDGE);
    float* nxt = ping[it & 1];
    k_gru<<<N_PATH / 16, 256, 0, stream>>>(agg, cur, gkt, grkt, gru_b, nxt);
    cur = nxt;
  }
  k_readout<<<N_PATH / 16, 256, 0, stream>>>(cur, R1t, rb1, R2, rb2, outp);
}
